// GraphAttentionLayer_54889682043153
// MI455X (gfx1250) — matrix operand order repaired and checked by reading, not compiled, not run on hardware
//
#include <hip/hip_runtime.h>
#include <math.h>

// ---------------- problem constants ----------------
constexpr int NN    = 864;    // nodes (54 * 16)
constexpr int IN_F  = 4096;   // 128 * 32
constexpr int OUT_F = 1024;   // 64 * 16
constexpr float ALPHA   = 0.2f;
constexpr float NEG_INF = -9e15f;

// ---------------- WMMA vector types ----------------
typedef __attribute__((ext_vector_type(16))) __bf16 bf16x16;
typedef __attribute__((ext_vector_type(8)))  float  f32x8;

struct alignas(16) U128 { unsigned int w[4]; };
union Frag { U128 u[2]; bf16x16 v; };

// ---------------- workspace layout (bytes) ----------------
constexpr size_t AL(size_t x) { return (x + 255) & ~size_t(255); }
constexpr size_t SZ_HBF   = (size_t)NN * IN_F * 2;      // h in bf16, row-major [NN][IN_F]
constexpr size_t SZ_WT    = (size_t)OUT_F * IN_F * 2;   // W^T in bf16, [OUT_F][IN_F]
constexpr size_t SZ_WH    = (size_t)NN * OUT_F * 4;     // Wh f32 [NN][OUT_F]
constexpr size_t SZ_WHT   = (size_t)OUT_F * NN * 2;     // Wh^T bf16 [OUT_F][NN]
constexpr size_t SZ_VEC   = 4096;                       // Wh1 / Wh2
constexpr size_t SZ_ATT   = (size_t)NN * NN * 4;        // att f32
constexpr size_t SZ_FUSED = (size_t)NN * NN * 2;        // fused bf16

constexpr size_t OFF_HBF   = 0;
constexpr size_t OFF_WT    = AL(OFF_HBF   + SZ_HBF);
constexpr size_t OFF_WH    = AL(OFF_WT    + SZ_WT);
constexpr size_t OFF_WHT   = AL(OFF_WH    + SZ_WH);
constexpr size_t OFF_WH1   = AL(OFF_WHT   + SZ_WHT);
constexpr size_t OFF_WH2   = AL(OFF_WH1   + SZ_VEC);
constexpr size_t OFF_ATT1  = AL(OFF_WH2   + SZ_VEC);
constexpr size_t OFF_ATT2  = AL(OFF_ATT1  + SZ_ATT);
constexpr size_t OFF_FUSED = AL(OFF_ATT2  + SZ_ATT);
constexpr size_t OFF_WGT   = AL(OFF_FUSED + SZ_FUSED);

// ---------------- helpers ----------------
__device__ __forceinline__ unsigned short f2bf(float f) {
    unsigned int u = __float_as_uint(f);
    unsigned int r = u + 0x7FFFu + ((u >> 16) & 1u);   // round-to-nearest-even
    return (unsigned short)(r >> 16);
}

__device__ __forceinline__ float block_reduce_sum(float v, float* sh) {
    int t = threadIdx.x;
    sh[t] = v; __syncthreads();
    for (int s = blockDim.x >> 1; s > 0; s >>= 1) {
        if (t < s) sh[t] += sh[t + s];
        __syncthreads();
    }
    float r = sh[0]; __syncthreads();
    return r;
}

__device__ __forceinline__ float block_reduce_max(float v, float* sh) {
    int t = threadIdx.x;
    sh[t] = v; __syncthreads();
    for (int s = blockDim.x >> 1; s > 0; s >>= 1) {
        if (t < s) sh[t] = fmaxf(sh[t], sh[t + s]);
        __syncthreads();
    }
    float r = sh[0]; __syncthreads();
    return r;
}

// ---------------- K0: h f32 -> bf16 ----------------
__global__ void k_cvt_h(const float* __restrict__ h, unsigned short* __restrict__ hbf) {
    int idx = blockIdx.x * blockDim.x + threadIdx.x;
    if (idx < NN * IN_F) hbf[idx] = f2bf(h[idx]);
}

// ---------------- K1: W [K][N] f32 -> W^T [N][K] bf16 (LDS-tiled) ----------------
__global__ void k_transpose_w(const float* __restrict__ W, unsigned short* __restrict__ Wt) {
    __shared__ unsigned short tile[32][33];
    int n     = blockIdx.x * 32 + threadIdx.x;
    int kbase = blockIdx.y * 32;
#pragma unroll
    for (int r = 0; r < 32; r += 8) {
        int k = kbase + threadIdx.y + r;
        tile[threadIdx.y + r][threadIdx.x] = f2bf(W[(size_t)k * OUT_F + n]);
    }
    __syncthreads();
#pragma unroll
    for (int r = 0; r < 32; r += 8) {
        int nOut = blockIdx.x * 32 + threadIdx.y + r;
        Wt[(size_t)nOut * IN_F + kbase + threadIdx.x] = tile[threadIdx.x][threadIdx.y + r];
    }
}

// ---------------- core: wave computes a 32x64 tile (2 M-subtiles x 4 N-tiles) ------
// A: bf16 row-major [M][lda]; B: bf16 N-major [N][ldb] (column of math-B contiguous).
// All 6 fragments of a k-step are loaded as fresh locals before the 8 WMMAs, so the
// scheduler can clause the loads and overlap next-iteration loads with WMMAs without
// exceeding the register budget (~130 VGPRs live).
__device__ __forceinline__ void wmma_strip2(
    const unsigned short* __restrict__ A, int lda,
    const unsigned short* __restrict__ B, int ldb,
    int mt2, int n_tile0, int ksteps, f32x8 (&acc)[2][4])
{
    const int lane = threadIdx.x & 31;
    const int half = lane >> 4;       // 0: lanes 0-15, 1: lanes 16-31
    const int l16  = lane & 15;

    const unsigned short* ar0 = A + (size_t)(mt2 * 32 + l16) * lda + half * 16;
    const unsigned short* ar1 = ar0 + (size_t)16 * lda;
    const unsigned short* bc0 = B + (size_t)((n_tile0 + 0) * 16 + l16) * ldb + half * 16;
    const unsigned short* bc1 = B + (size_t)((n_tile0 + 1) * 16 + l16) * ldb + half * 16;
    const unsigned short* bc2 = B + (size_t)((n_tile0 + 2) * 16 + l16) * ldb + half * 16;
    const unsigned short* bc3 = B + (size_t)((n_tile0 + 3) * 16 + l16) * ldb + half * 16;

    for (int kt = 0; kt < ksteps; ++kt) {
        const int kk = kt * 32;
        // A fragment layout per lane: K {0..15} (lanes 0-15) / {16..31}, contiguous
        Frag a0, a1, b0, b1, b2, b3;
        a0.u[0] = *reinterpret_cast<const U128*>(ar0 + kk);
        a0.u[1] = *reinterpret_cast<const U128*>(ar0 + kk + 8);
        a1.u[0] = *reinterpret_cast<const U128*>(ar1 + kk);
        a1.u[1] = *reinterpret_cast<const U128*>(ar1 + kk + 8);
        // B fragment layout per lane: K {0..15} (lanes 0-15) / {16..31}, contiguous
        b0.u[0] = *reinterpret_cast<const U128*>(bc0 + kk);
        b0.u[1] = *reinterpret_cast<const U128*>(bc0 + kk + 8);
        b1.u[0] = *reinterpret_cast<const U128*>(bc1 + kk);
        b1.u[1] = *reinterpret_cast<const U128*>(bc1 + kk + 8);
        b2.u[0] = *reinterpret_cast<const U128*>(bc2 + kk);
        b2.u[1] = *reinterpret_cast<const U128*>(bc2 + kk + 8);
        b3.u[0] = *reinterpret_cast<const U128*>(bc3 + kk);
        b3.u[1] = *reinterpret_cast<const U128*>(bc3 + kk + 8);
        if (kt + 1 < ksteps)
            __builtin_prefetch(ar0 + kk + 32, 0, 1);   // global_prefetch next k-step

        acc[0][0] = __builtin_amdgcn_wmma_f32_16x16x32_bf16(false, a0.v, false, b0.v, (short)0, acc[0][0], false, false);
        acc[0][1] = __builtin_amdgcn_wmma_f32_16x16x32_bf16(false, a0.v, false, b1.v, (short)0, acc[0][1], false, false);
        acc[0][2] = __builtin_amdgcn_wmma_f32_16x16x32_bf16(false, a0.v, false, b2.v, (short)0, acc[0][2], false, false);
        acc[0][3] = __builtin_amdgcn_wmma_f32_16x16x32_bf16(false, a0.v, false, b3.v, (short)0, acc[0][3], false, false);
        acc[1][0] = __builtin_amdgcn_wmma_f32_16x16x32_bf16(false, a1.v, false, b0.v, (short)0, acc[1][0], false, false);
        acc[1][1] = __builtin_amdgcn_wmma_f32_16x16x32_bf16(false, a1.v, false, b1.v, (short)0, acc[1][1], false, false);
        acc[1][2] = __builtin_amdgcn_wmma_f32_16x16x32_bf16(false, a1.v, false, b2.v, (short)0, acc[1][2], false, false);
        acc[1][3] = __builtin_amdgcn_wmma_f32_16x16x32_bf16(false, a1.v, false, b3.v, (short)0, acc[1][3], false, false);
    }
}

// ---------------- K2: Wh = h @ W  (bf16 WMMA, f32 accum) ----------------
// 8 waves/block, wave -> (32-row m-pair, 4 consecutive n-tiles). 54 blocks = 432 waves.
__global__ void k_gemm1(const unsigned short* __restrict__ hbf,
                        const unsigned short* __restrict__ wt,
                        float* __restrict__ Wh,
                        unsigned short* __restrict__ WhT)
{
    const int w   = blockIdx.x * 8 + (threadIdx.x >> 5);
    const int mt2 = w >> 4;         // 0..26  (pair of 16-row tiles)
    const int ns  = w & 15;         // 0..15  -> n tiles ns*4 .. ns*4+3
    const f32x8 zero = {0.f,0.f,0.f,0.f,0.f,0.f,0.f,0.f};
    f32x8 acc[2][4] = {{zero, zero, zero, zero}, {zero, zero, zero, zero}};

    wmma_strip2(hbf, IN_F, wt, IN_F, mt2, ns * 4, IN_F / 32, acc);

    const int lane = threadIdx.x & 31;
    const int half = lane >> 4;
    const int l16  = lane & 15;
#pragma unroll
    for (int i = 0; i < 2; ++i) {
#pragma unroll
        for (int j = 0; j < 4; ++j) {
            const int n = (ns * 4 + j) * 16 + l16;
#pragma unroll
            for (int r = 0; r < 8; ++r) {
                const int m = mt2 * 32 + i * 16 + r + half * 8;  // C/D: VGPR r -> M r / r+8
                const float v = acc[i][j][r];
                Wh[(size_t)m * OUT_F + n] = v;
                WhT[(size_t)n * NN + m]   = f2bf(v);
            }
        }
    }
}

// ---------------- K3: Wh1 = Wh @ a[:F], Wh2 = Wh @ a[F:] ----------------
__global__ void k_gemv(const float* __restrict__ Wh, const float* __restrict__ a,
                       float* __restrict__ Wh1, float* __restrict__ Wh2)
{
    __shared__ float s1[256];
    __shared__ float s2[256];
    const int i = blockIdx.x;
    float p1 = 0.f, p2 = 0.f;
    for (int c = threadIdx.x; c < OUT_F; c += 256) {
        const float v = Wh[(size_t)i * OUT_F + c];
        p1 += v * a[c];
        p2 += v * a[c + OUT_F];
    }
    const float r1 = block_reduce_sum(p1, s1);
    const float r2 = block_reduce_sum(p2, s2);
    if (threadIdx.x == 0) { Wh1[i] = r1; Wh2[i] = r2; }
}

// ---------------- K4: masked row-softmax for both adjacency maps ----------------
__global__ void k_attention(const float* __restrict__ Wh1, const float* __restrict__ Wh2,
                            const int* __restrict__ adj1, const int* __restrict__ adj2,
                            float* __restrict__ att1, float* __restrict__ att2)
{
    __shared__ float sh[256];
    const int i = blockIdx.x;
    const float wi = Wh1[i];

    float m1 = NEG_INF, m2 = NEG_INF;
    for (int j = threadIdx.x; j < NN; j += 256) {
        const float s = wi + Wh2[j];
        const float e = s > 0.f ? s : ALPHA * s;
        const float l1 = adj1[(size_t)i * NN + j] > 0 ? e : NEG_INF;
        const float l2 = adj2[(size_t)i * NN + j] > 0 ? e : NEG_INF;
        m1 = fmaxf(m1, l1);
        m2 = fmaxf(m2, l2);
    }
    const float M1 = block_reduce_max(m1, sh);
    const float M2 = block_reduce_max(m2, sh);

    float su1 = 0.f, su2 = 0.f;
    for (int j = threadIdx.x; j < NN; j += 256) {
        const float s = wi + Wh2[j];
        const float e = s > 0.f ? s : ALPHA * s;
        const float l1 = adj1[(size_t)i * NN + j] > 0 ? e : NEG_INF;
        const float l2 = adj2[(size_t)i * NN + j] > 0 ? e : NEG_INF;
        su1 += __expf(l1 - M1);
        su2 += __expf(l2 - M2);
    }
    const float S1 = block_reduce_sum(su1, sh);
    const float S2 = block_reduce_sum(su2, sh);
    const float inv1 = 1.f / S1, inv2 = 1.f / S2;

    for (int j = threadIdx.x; j < NN; j += 256) {
        const float s = wi + Wh2[j];
        const float e = s > 0.f ? s : ALPHA * s;
        const float l1 = adj1[(size_t)i * NN + j] > 0 ? e : NEG_INF;
        const float l2 = adj2[(size_t)i * NN + j] > 0 ? e : NEG_INF;
        att1[(size_t)i * NN + j] = __expf(l1 - M1) * inv1;
        att2[(size_t)i * NN + j] = __expf(l2 - M2) * inv2;
    }
}

// ---------------- K5: 16x16 maxpool -> linear -> leaky -> 2-way softmax ----------
__global__ void k_fusion_weights(const float* __restrict__ att1, const float* __restrict__ att2,
                                 const float* __restrict__ L_w, const float* __restrict__ L_b,
                                 float* __restrict__ wgt)
{
    __shared__ float sh[1024];
    float s0 = 0.f, s1 = 0.f;
    for (int c = threadIdx.x; c < 2 * 2916; c += 1024) {
        const int map  = c / 2916;
        const int cell = c % 2916;
        const int pr = cell / 54, pc = cell % 54;
        const float* att = map ? att2 : att1;
        float v = -1e30f;
        for (int r = 0; r < 16; ++r) {
            const float* rowp = att + (size_t)(pr * 16 + r) * NN + pc * 16;
#pragma unroll
            for (int cc = 0; cc < 16; ++cc) v = fmaxf(v, rowp[cc]);
        }
        const float contrib = v * L_w[cell];
        if (map) s1 += contrib; else s0 += contrib;
    }
    const float t0 = block_reduce_sum(s0, sh);
    const float t1 = block_reduce_sum(s1, sh);
    if (threadIdx.x == 0) {
        const float b = L_b[0];
        float l0 = t0 + b, l1 = t1 + b;
        l0 = l0 > 0.f ? l0 : ALPHA * l0;
        l1 = l1 > 0.f ? l1 : ALPHA * l1;
        const float m  = fmaxf(l0, l1);
        const float e0 = __expf(l0 - m), e1 = __expf(l1 - m);
        wgt[0] = e0 / (e0 + e1);
        wgt[1] = e1 / (e0 + e1);
    }
}

// ---------------- K6: fused = w0*att1 + w1*att2 -> bf16 ----------------
__global__ void k_fuse(const float* __restrict__ att1, const float* __restrict__ att2,
                       const float* __restrict__ wgt, unsigned short* __restrict__ fused)
{
    const int idx = blockIdx.x * blockDim.x + threadIdx.x;
    if (idx < NN * NN) {
        const float f = att1[idx] * wgt[0] + att2[idx] * wgt[1];
        fused[idx] = f2bf(f);
    }
}

// ---------------- K7: out = elu(fused @ Wh)  (bf16 WMMA, f32 accum) ----------------
__global__ void k_gemm2(const unsigned short* __restrict__ fusedbf,
                        const unsigned short* __restrict__ whT,
                        float* __restrict__ out)
{
    const int w   = blockIdx.x * 8 + (threadIdx.x >> 5);
    const int mt2 = w >> 4;
    const int ns  = w & 15;
    const f32x8 zero = {0.f,0.f,0.f,0.f,0.f,0.f,0.f,0.f};
    f32x8 acc[2][4] = {{zero, zero, zero, zero}, {zero, zero, zero, zero}};

    wmma_strip2(fusedbf, NN, whT, NN, mt2, ns * 4, NN / 32, acc);   // 27 k-steps

    const int lane = threadIdx.x & 31;
    const int half = lane >> 4;
    const int l16  = lane & 15;
#pragma unroll
    for (int i = 0; i < 2; ++i) {
#pragma unroll
        for (int j = 0; j < 4; ++j) {
            const int n = (ns * 4 + j) * 16 + l16;
#pragma unroll
            for (int r = 0; r < 8; ++r) {
                const int m = mt2 * 32 + i * 16 + r + half * 8;
                const float x = acc[i][j][r];
                out[(size_t)m * OUT_F + n] = x > 0.f ? x : (__expf(x) - 1.f);
            }
        }
    }
}

// ---------------- launcher ----------------
extern "C" void kernel_launch(void* const* d_in, const int* in_sizes, int n_in,
                              void* d_out, int out_size, void* d_ws, size_t ws_size,
                              hipStream_t stream)
{
    const float* h    = (const float*)d_in[0];
    // d_in[1] ("adj") is unused by the reference
    const int*   adj1 = (const int*)d_in[2];
    const int*   adj2 = (const int*)d_in[3];
    const float* W    = (const float*)d_in[4];
    const float* a    = (const float*)d_in[5];
    const float* L_w  = (const float*)d_in[6];
    const float* L_b  = (const float*)d_in[7];
    float* out = (float*)d_out;

    char* ws = (char*)d_ws;
    unsigned short* hbf   = (unsigned short*)(ws + OFF_HBF);
    unsigned short* wt    = (unsigned short*)(ws + OFF_WT);
    float*          Wh    = (float*)(ws + OFF_WH);
    unsigned short* WhT   = (unsigned short*)(ws + OFF_WHT);
    float*          Wh1   = (float*)(ws + OFF_WH1);
    float*          Wh2   = (float*)(ws + OFF_WH2);
    float*          att1  = (float*)(ws + OFF_ATT1);
    float*          att2  = (float*)(ws + OFF_ATT2);
    unsigned short* fused = (unsigned short*)(ws + OFF_FUSED);
    float*          wgt   = (float*)(ws + OFF_WGT);

    k_cvt_h      <<<(NN * IN_F) / 256, 256, 0, stream>>>(h, hbf);
    k_transpose_w<<<dim3(OUT_F / 32, IN_F / 32), dim3(32, 8), 0, stream>>>(W, wt);
    k_gemm1      <<<(27 * 16) / 8, 256, 0, stream>>>(hbf, wt, Wh, WhT);
    k_gemv       <<<NN, 256, 0, stream>>>(Wh, a, Wh1, Wh2);
    k_attention  <<<NN, 256, 0, stream>>>(Wh1, Wh2, adj1, adj2, att1, att2);
    k_fusion_weights<<<1, 1024, 0, stream>>>(att1, att2, L_w, L_b, wgt);
    k_fuse       <<<(NN * NN) / 256, 256, 0, stream>>>(att1, att2, wgt, fused);
    k_gemm2      <<<(27 * 16) / 8, 256, 0, stream>>>(fused, WhT, out);
}
